// MP_GNN_15178414424332
// MI455X (gfx1250) — matrix operand order repaired and checked by reading, not compiled, not run on hardware
//
#include <hip/hip_runtime.h>
#include <utility>

// ---------------------------------------------------------------------------
// MP-GNN for MI455X (gfx1250, wave32). All GEMMs via v_wmma_f32_16x16x32_f16.
// Activations stored f16 (halves L2/HBM gather traffic), accum f32.
// ---------------------------------------------------------------------------

typedef __attribute__((ext_vector_type(16))) _Float16 v16h;
typedef __attribute__((ext_vector_type(8)))  _Float16 v8h;
typedef __attribute__((ext_vector_type(8)))  float    v8f;

__device__ inline v8f v8f_zero() {
    v8f z;
#pragma unroll
    for (int i = 0; i < 8; ++i) z[i] = 0.f;
    return z;
}

// A-operand (16x32 f16, MxK) lane layout:
//   lanes 0-15 : row = lane,    K = {0..7, 16..23}
//   lanes 16-31: row = lane-16, K = {8..15, 24..31}
// p = row base (f16, row-major, +kc*32 already applied)
__device__ inline v16h load_a16(const _Float16* p, int hi) {
    union { v16h v; v8h h[2]; } u;
    u.h[0] = *(const v8h*)(p + hi * 8);
    u.h[1] = *(const v8h*)(p + 16 + hi * 8);
    return u.v;
}

// Build padded A tile from a narrow f32 row (C = 7 or 3 real columns).
__device__ inline v16h load_a_pad(const float* rowp, int C, int hi) {
    union { v16h v; _Float16 e[16]; } u;
#pragma unroll
    for (int j = 0; j < 16; ++j) {
        int k = j + ((j >> 3) << 3) + hi * 8;   // j<8 -> j+8*hi ; j>=8 -> j+8+8*hi
        u.e[j] = (k < C) ? (_Float16)rowp[k] : (_Float16)0.f;
    }
    return u.v;
}

struct MlpArgs {
    const float*    srcF;    // ENC: raw f32 features
    const _Float16* x16;     // node features (f16, [*,64])
    const _Float16* e16;     // edge features in (f16, [*,64])
    const _Float16* agg16;   // aggregated node features (f16, [*,64])
    const int*      eidx;    // edge_index flat [2*E]: src=[0,E), dst=[E,2E)
    const _Float16* W1p;     // packed layer-1 weights (B-operand layout)
    const float*    b1;
    const _Float16* W2p;     // packed layer-2 weights
    const float*    b2;
    const float*    g;       // LN gamma
    const float*    bt;      // LN beta
    _Float16*       out16;
    float*          out32;
    float*          agg;     // f32 scatter accumulator (EDGE mode)
    int rows, srcC, nE;
};

// MODE: 0 = encode (f32 padded input, K=32)  1 = edge MP (K=192 gathered)
//       2 = node MP (K=128)                  3 = decode (K=64, out f32, no LN)
template <int MODE>
__global__ __launch_bounds__(256) void mlp2_kernel(MlpArgs A) {
    constexpr int KCH1 = (MODE == 0) ? 1 : (MODE == 1) ? 6 : (MODE == 2) ? 4 : 2;
    constexpr int NT2  = (MODE == 3) ? 1 : 4;

    __shared__ __align__(32) _Float16 lds[8][16 * 64];

    const int lane = threadIdx.x & 31;
    const int wave = threadIdx.x >> 5;
    const int tile = blockIdx.x * 8 + wave;
    const long long row0 = (long long)tile * 16;
    if (row0 >= A.rows) return;

    const int hi = lane >> 4;       // which K-half this lane carries for A
    const int c  = lane & 15;       // column-in-tile for B/C/D, row-in-tile for A
    const long long rowA = row0 + c;
    const long long rA   = (rowA < A.rows) ? rowA : 0;

    // ---- layer 1: acc1 = A @ W1 -------------------------------------------
    v8f acc1[4];
#pragma unroll
    for (int t = 0; t < 4; ++t) acc1[t] = v8f_zero();

    const _Float16* ax = nullptr;
    const _Float16* ay = nullptr;
    const _Float16* ae = nullptr;
    if constexpr (MODE == 1) {
        int s = A.eidx[rA];
        int d = A.eidx[(long long)A.nE + rA];
        ax = A.x16 + (long long)s * 64;
        ay = A.x16 + (long long)d * 64;
        ae = A.e16 + rA * 64;
    } else if constexpr (MODE == 2) {
        ax = A.x16 + rA * 64;
        ay = A.agg16 + rA * 64;
    } else if constexpr (MODE == 3) {
        ax = A.x16 + rA * 64;
    }

#pragma unroll
    for (int kc = 0; kc < KCH1; ++kc) {
        v16h a;
        if constexpr (MODE == 0) {
            a = load_a_pad(A.srcF + rA * A.srcC, A.srcC, hi);
        } else if constexpr (MODE == 1) {
            const _Float16* p = (kc < 2) ? ax + kc * 32
                              : (kc < 4) ? ay + (kc - 2) * 32
                                         : ae + (kc - 4) * 32;
            a = load_a16(p, hi);
        } else if constexpr (MODE == 2) {
            const _Float16* p = (kc < 2) ? ax + kc * 32 : ay + (kc - 2) * 32;
            a = load_a16(p, hi);
        } else {
            a = load_a16(ax + kc * 32, hi);
        }
#pragma unroll
        for (int t = 0; t < 4; ++t) {
            v16h b = *(const v16h*)(A.W1p + ((size_t)(kc * 4 + t)) * 512 + lane * 16);
            acc1[t] = __builtin_amdgcn_wmma_f32_16x16x32_f16(
                false, a, false, b, (short)0, acc1[t], false, false);
        }
    }

    // ---- bias + ELU, stage f16 row-major in LDS ---------------------------
    // D layout: vgpr v -> row v + 8*hi ; col = t*16 + c
    _Float16* hst = lds[wave];
#pragma unroll
    for (int t = 0; t < 4; ++t) {
        float bb = A.b1[t * 16 + c];
#pragma unroll
        for (int v = 0; v < 8; ++v) {
            float xv = acc1[t][v] + bb;
            xv = xv > 0.f ? xv : (__expf(xv) - 1.f);   // ELU
            hst[(v + hi * 8) * 64 + t * 16 + c] = (_Float16)xv;
        }
    }
    // intra-wave LDS store->load: DS ops are in-order per wave

    // ---- layer 2: K = 64 --------------------------------------------------
    v8f acc2[NT2];
#pragma unroll
    for (int t = 0; t < NT2; ++t) acc2[t] = v8f_zero();
#pragma unroll
    for (int kc = 0; kc < 2; ++kc) {
        v16h a = load_a16(hst + c * 64 + kc * 32, hi);
#pragma unroll
        for (int t = 0; t < NT2; ++t) {
            v16h b = *(const v16h*)(A.W2p + ((size_t)(kc * NT2 + t)) * 512 + lane * 16);
            acc2[t] = __builtin_amdgcn_wmma_f32_16x16x32_f16(
                false, a, false, b, (short)0, acc2[t], false, false);
        }
    }

    float vals[4][8];
#pragma unroll
    for (int t = 0; t < NT2; ++t) {
        float bb;
        if constexpr (MODE == 3) bb = (c < 7) ? A.b2[c] : 0.f;
        else                     bb = A.b2[t * 16 + c];
#pragma unroll
        for (int v = 0; v < 8; ++v) vals[t][v] = acc2[t][v] + bb;
    }

    if constexpr (MODE == 3) {  // decode: write f32 [rows,7], no LayerNorm
#pragma unroll
        for (int v = 0; v < 8; ++v) {
            long long m = row0 + v + hi * 8;
            if (m < A.rows && c < 7) A.out32[m * 7 + c] = vals[0][v];
        }
        return;
    }

    // ---- LayerNorm over 64 cols; rows live across the 16-lane half-group --
#pragma unroll
    for (int v = 0; v < 8; ++v) {
        float s = 0.f, q = 0.f;
#pragma unroll
        for (int t = 0; t < 4; ++t) { float xv = vals[t][v]; s += xv; q += xv * xv; }
#pragma unroll
        for (int off = 8; off >= 1; off >>= 1) {
            s += __shfl_xor(s, off, 32);
            q += __shfl_xor(q, off, 32);
        }
        float mean = s * (1.f / 64.f);
        float var  = q * (1.f / 64.f) - mean * mean;
        float rstd = rsqrtf(var + 1e-5f);

        long long m = row0 + v + hi * 8;
        bool ok = (m < A.rows);
        long long dIdx = 0;
        if constexpr (MODE == 1) { if (ok) dIdx = A.eidx[(long long)A.nE + m]; }
#pragma unroll
        for (int t = 0; t < 4; ++t) {
            float yv = (vals[t][v] - mean) * rstd * A.g[t * 16 + c] + A.bt[t * 16 + c];
            if (ok) {
                A.out16[m * 64 + t * 16 + c] = (_Float16)yv;
                if constexpr (MODE == 1)
                    atomicAdd(&A.agg[dIdx * 64 + t * 16 + c], yv);  // scatter-mean numerator
            }
        }
    }
}

// ---------------------------------------------------------------------------
// Helper kernels
// ---------------------------------------------------------------------------

// Pack f32 weight [K,N] into WMMA B-operand lane layout: per 32x16 tile, lane L
// holds 16 contiguous halfs = column (L&15), K-locals ((j>>3)*16 + (L>>4)*8 + (j&7)).
__global__ void pack_kernel(const float* W, _Float16* out, int K, int N, int KC, int NT) {
    long long idx = (long long)blockIdx.x * blockDim.x + threadIdx.x;
    long long total = (long long)KC * NT * 512;
    if (idx >= total) return;
    int t  = (int)(idx >> 9);
    int r  = (int)(idx & 511);
    int L  = r >> 4, j = r & 15;
    int kc = t / NT, nc = t % NT;
    int k  = kc * 32 + ((j >> 3) << 4) + ((L >> 4) << 3) + (j & 7);
    int n  = nc * 16 + (L & 15);
    out[idx] = (k < K && n < N) ? (_Float16)W[(long long)k * N + n] : (_Float16)0.f;
}

__global__ void zero_kernel(float* p, long long n) {
    long long i = (long long)blockIdx.x * blockDim.x + threadIdx.x;
    if (i < n) p[i] = 0.f;
}

__global__ void count_kernel(const int* eidx, int E, float* cnt) {
    int i = blockIdx.x * blockDim.x + threadIdx.x;
    if (i < E) atomicAdd(&cnt[eidx[E + i]], 1.0f);
}

__global__ void aggdiv_kernel(const float* agg, const float* cnt, _Float16* out, long long total) {
    long long i = (long long)blockIdx.x * blockDim.x + threadIdx.x;
    if (i >= total) return;
    float d = cnt[i >> 6];
    if (d < 1.f) d = 1.f;
    out[i] = (_Float16)(agg[i] / d);
}

// ---------------------------------------------------------------------------
// Host orchestration
// ---------------------------------------------------------------------------
extern "C" void kernel_launch(void* const* d_in, const int* in_sizes, int n_in,
                              void* d_out, int out_size, void* d_ws, size_t ws_size,
                              hipStream_t stream) {
    (void)in_sizes; (void)n_in; (void)out_size; (void)ws_size;
    const int N = 100000, E = 1600000;

    const float* x     = (const float*)d_in[0];
    const int*   eidx  = (const int*)d_in[1];   // jnp.int64 -> int32 (JAX x64 off)
    const float* eattr = (const float*)d_in[2];
    auto F = [&](int i) { return (const float*)d_in[i]; };

    char* wsp = (char*)d_ws;
    auto carve = [&](size_t bytes) -> char* {
        char* p = wsp; wsp += (bytes + 255) & ~(size_t)255; return p;
    };

    _Float16* x16a  = (_Float16*)carve((size_t)N * 64 * 2);
    _Float16* x16b  = (_Float16*)carve((size_t)N * 64 * 2);
    _Float16* e16a  = (_Float16*)carve((size_t)E * 64 * 2);
    _Float16* e16b  = (_Float16*)carve((size_t)E * 64 * 2);
    float*    agg   = (float*)   carve((size_t)N * 64 * 4);
    _Float16* agg16 = (_Float16*)carve((size_t)N * 64 * 2);
    float*    cnt   = (float*)   carve((size_t)N * 4);

    // Pack every weight matrix into WMMA B-operand layout.
    struct PD { int in, K, N; };
    const PD pd[14] = {
        {3, 7, 64},  {5, 64, 64},            // node encode W1,W2
        {9, 3, 64},  {11, 64, 64},           // edge encode W1,W2
        {15, 192, 64}, {17, 64, 64},         // edge MP 0
        {19, 192, 64}, {21, 64, 64},         // edge MP 1
        {27, 128, 64}, {29, 64, 64},         // node MP 0
        {31, 128, 64}, {33, 64, 64},         // node MP 1
        {39, 64, 64}, {41, 64, 7},           // decode W1,W2
    };
    _Float16* pw[14];
    for (int i = 0; i < 14; ++i) {
        int KC = (pd[i].K + 31) / 32, NT = (pd[i].N + 15) / 16;
        size_t elems = (size_t)KC * NT * 512;
        pw[i] = (_Float16*)carve(elems * 2);
        pack_kernel<<<(unsigned)((elems + 255) / 256), 256, 0, stream>>>(
            F(pd[i].in), pw[i], pd[i].K, pd[i].N, KC, NT);
    }

    // in-degree counts (scatter-mean denominator)
    zero_kernel<<<(N + 255) / 256, 256, 0, stream>>>(cnt, N);
    count_kernel<<<(E + 255) / 256, 256, 0, stream>>>(eidx, E, cnt);

    // encode nodes
    {
        MlpArgs a{}; a.srcF = x; a.srcC = 7;
        a.W1p = pw[0]; a.b1 = F(4); a.W2p = pw[1]; a.b2 = F(6);
        a.g = F(7); a.bt = F(8); a.out16 = x16a; a.rows = N;
        mlp2_kernel<0><<<(N + 127) / 128, 256, 0, stream>>>(a);
    }
    // encode edges
    {
        MlpArgs a{}; a.srcF = eattr; a.srcC = 3;
        a.W1p = pw[2]; a.b1 = F(10); a.W2p = pw[3]; a.b2 = F(12);
        a.g = F(13); a.bt = F(14); a.out16 = e16a; a.rows = E;
        mlp2_kernel<0><<<(E + 127) / 128, 256, 0, stream>>>(a);
    }

    _Float16 *xc = x16a, *xn = x16b, *ec = e16a, *en = e16b;
    for (int i = 0; i < 2; ++i) {
        zero_kernel<<<(unsigned)(((size_t)N * 64 + 255) / 256), 256, 0, stream>>>(agg, (long long)N * 64);

        // edge update: concat(x[src], x[dst], e) -> MLP -> LN (+ fused scatter)
        MlpArgs e{}; e.x16 = xc; e.e16 = ec; e.eidx = eidx; e.nE = E;
        e.W1p = pw[i ? 6 : 4]; e.b1 = F(i ? 20 : 16);
        e.W2p = pw[i ? 7 : 5]; e.b2 = F(i ? 22 : 18);
        e.g = F(i ? 25 : 23); e.bt = F(i ? 26 : 24);
        e.out16 = en; e.agg = agg; e.rows = E;
        mlp2_kernel<1><<<(E + 127) / 128, 256, 0, stream>>>(e);

        aggdiv_kernel<<<(unsigned)(((size_t)N * 64 + 255) / 256), 256, 0, stream>>>(
            agg, cnt, agg16, (long long)N * 64);

        // node update: concat(x, agg) -> MLP -> LN
        MlpArgs nA{}; nA.x16 = xc; nA.agg16 = agg16;
        nA.W1p = pw[i ? 10 : 8]; nA.b1 = F(i ? 32 : 28);
        nA.W2p = pw[i ? 11 : 9]; nA.b2 = F(i ? 34 : 30);
        nA.g = F(i ? 37 : 35); nA.bt = F(i ? 38 : 36);
        nA.out16 = xn; nA.rows = N;
        mlp2_kernel<2><<<(N + 127) / 128, 256, 0, stream>>>(nA);

        std::swap(xc, xn); std::swap(ec, en);
    }

    // decode -> f32 output [N,7]
    {
        MlpArgs d{}; d.x16 = xc;
        d.W1p = pw[12]; d.b1 = F(40); d.W2p = pw[13]; d.b2 = F(42);
        d.out32 = (float*)d_out; d.rows = N;
        mlp2_kernel<3><<<(N + 127) / 128, 256, 0, stream>>>(d);
    }
}
